// WeightSharedLinear_30313879175885
// MI455X (gfx1250) — matrix operand order repaired and checked by reading, not compiled, not run on hardware
//
#include <hip/hip_runtime.h>
#include <hip/hip_bf16.h>

// Problem constants (fixed by the reference).
#define BATCH   4
#define SEQ     2048
#define MDIM    (BATCH * SEQ)   // 8192 rows of x
#define KDIM    4096            // inner (IN)
#define NDIM    4096            // OUT
#define NCENT   256

// GEMM tiling.
#define BLK_M   128             // block tile M
#define BLK_N   128             // block tile N
#define BLK_K   64              // K stage depth
#define WAVE_M  64              // per-wave tile M (2x2 waves per block)
#define WAVE_N  64              // per-wave tile N
// LDS: TDM pads 16B after every 128B row -> row stride 144B (36 banks).
#define LDS_ROW_BYTES   144
#define LDS_ROW_ELEMS   72
#define LDS_TILE_BYTES  (BLK_M * LDS_ROW_BYTES)   // 18432 per tile
#define LDS_TOTAL       (4 * LDS_TILE_BYTES)      // A[2 stages] + B[2 stages] = 73728

typedef __attribute__((ext_vector_type(16))) __bf16 bf16x16;
typedef __attribute__((ext_vector_type(8)))  __bf16 bf16x8;
typedef __attribute__((ext_vector_type(8)))  float  f32x8;
typedef __attribute__((ext_vector_type(4)))  unsigned int u32x4;
typedef __attribute__((ext_vector_type(4)))  int    i32x4;
typedef __attribute__((ext_vector_type(8)))  int    i32x8;

union Bf16Frag {
    bf16x16 v;
    bf16x8  h[2];
};

#if __has_builtin(__builtin_amdgcn_tensor_load_to_lds) && __has_builtin(__builtin_amdgcn_s_wait_tensorcnt)
#define HAS_TDM 1
#else
#define HAS_TDM 0
#endif

// ---------------------------------------------------------------------------
// Kernel 1: dequantize weight codes -> bf16 weight matrix in workspace.
// ---------------------------------------------------------------------------
__global__ void wsl_dequant_weight(const long long* __restrict__ indices,
                                   const float* __restrict__ centroids,
                                   __bf16* __restrict__ wb) {
    __shared__ __bf16 lut[NCENT];
    int tid = threadIdx.x;
    if (tid < NCENT) lut[tid] = (__bf16)centroids[tid];
    __syncthreads();

    size_t e = ((size_t)blockIdx.x * blockDim.x + tid) * 8;
    const ulonglong2* ip = (const ulonglong2*)(indices + e);
    ulonglong2 i0 = ip[0], i1 = ip[1], i2 = ip[2], i3 = ip[3];

    bf16x8 o;
    o[0] = lut[(int)i0.x & 255]; o[1] = lut[(int)i0.y & 255];
    o[2] = lut[(int)i1.x & 255]; o[3] = lut[(int)i1.y & 255];
    o[4] = lut[(int)i2.x & 255]; o[5] = lut[(int)i2.y & 255];
    o[6] = lut[(int)i3.x & 255]; o[7] = lut[(int)i3.y & 255];
    *(bf16x8*)(wb + e) = o;
}

// ---------------------------------------------------------------------------
// Kernel 2: convert activations fp32 -> bf16.
// ---------------------------------------------------------------------------
__global__ void wsl_convert_x(const float* __restrict__ x,
                              __bf16* __restrict__ xb) {
    size_t e = ((size_t)blockIdx.x * blockDim.x + threadIdx.x) * 8;
    const float4* xp = (const float4*)(x + e);
    float4 a = xp[0];
    float4 b = xp[1];
    bf16x8 o;
    o[0] = (__bf16)a.x; o[1] = (__bf16)a.y; o[2] = (__bf16)a.z; o[3] = (__bf16)a.w;
    o[4] = (__bf16)b.x; o[5] = (__bf16)b.y; o[6] = (__bf16)b.z; o[7] = (__bf16)b.w;
    *(bf16x8*)(xb + e) = o;
}

// ---------------------------------------------------------------------------
// TDM issue: DMA one 128-row x 64-element bf16 tile (K-contiguous rows,
// tensor row stride = KDIM elems) into LDS at ldsOff, padding 16B after
// every 128B row (pad_interval code 4 = 32 DWORDs, pad_amount code 3 = 4
// DWORDs). Tracked by TENSORcnt. Wave-level op: call from one wave.
// ---------------------------------------------------------------------------
#if HAS_TDM
__device__ __forceinline__ void tdm_load_tile(const __bf16* gsrc,
                                              unsigned ldsOff,
                                              unsigned tensorRows) {
    unsigned long long ga = (unsigned long long)gsrc;
    u32x4 g0;
    g0[0] = 1u;                                   // count=1, load, user D#
    g0[1] = ldsOff;                               // lds_addr (bytes)
    g0[2] = (unsigned)ga;                         // global_addr[31:0]
    g0[3] = (unsigned)((ga >> 32) & 0x01FFFFFFu)  // global_addr[56:32]
          | (2u << 30);                           // type=2 (image)
    i32x8 g1;
    g1[0] = (int)((1u << 16)                      // data_size = 2B
                | (1u << 20)                      // pad_enable
                | (4u << 22)                      // pad_interval: 32 DWORDs
                | (3u << 25));                    // pad_amount: 4 DWORDs (16B)
    g1[1] = (int)((unsigned)KDIM << 16);          // tensor_dim0[15:0]
    g1[2] = (int)(((unsigned)KDIM >> 16)          // tensor_dim0[31:16]
                | ((tensorRows & 0xFFFFu) << 16));// tensor_dim1[15:0]
    g1[3] = (int)((tensorRows >> 16)              // tensor_dim1[31:16]
                | ((unsigned)BLK_K << 16));       // tile_dim0 = 64
    g1[4] = (int)BLK_M;                           // tile_dim1 = 128, tile_dim2 = 0
    g1[5] = (int)KDIM;                            // tensor_dim0_stride[31:0]
    g1[6] = 0;                                    // stride hi / dim1_stride lo
    g1[7] = 0;
    i32x4 z4 = (i32x4){0, 0, 0, 0};
#if defined(__clang_major__) && (__clang_major__ >= 23)
    i32x8 z8 = (i32x8){0, 0, 0, 0, 0, 0, 0, 0};
    __builtin_amdgcn_tensor_load_to_lds(g0, g1, z4, z4, z8, 0);
#else
    __builtin_amdgcn_tensor_load_to_lds(g0, g1, z4, z4, 0);
#endif
}
#else
// Fallback: cooperative copy, one row per thread, same padded LDS layout.
__device__ __forceinline__ void copy_tile_fallback(const __bf16* __restrict__ gsrc,
                                                   __bf16* __restrict__ dst,
                                                   int tid) {
    const __bf16* s = gsrc + (size_t)tid * KDIM;
    __bf16* d = dst + tid * LDS_ROW_ELEMS;
#pragma unroll
    for (int c = 0; c < BLK_K; c += 8)
        *(bf16x8*)(d + c) = *(const bf16x8*)(s + c);
}
#endif

// ---------------------------------------------------------------------------
// Kernel 3: bf16 WMMA GEMM, TDM-staged LDS double buffering.
//   Block = 128 threads = 4 waves (2x2), block tile 128x128, wave tile 64x64.
//   Stage = 64 K. Per 32-K sub-step: 16 ds_load_b128 + 16 v_wmma per wave.
//
// NOTE: no aggregates may be initialized with the __shared__ symbol (the
// addrspacecast is not a link-time constant) — all LDS addressing below is
// scalar byte-offset arithmetic done at runtime.
// ---------------------------------------------------------------------------
__global__ void __launch_bounds__(128, 1)
wsl_wmma_gemm(const __bf16* __restrict__ xb,
              const __bf16* __restrict__ wb,
              const float* __restrict__ bias,
              float* __restrict__ out) {
    __shared__ __align__(16) unsigned char smem[LDS_TOTAL];

    const int tid    = threadIdx.x;
    const int lane   = tid & 31;
    const int waveId = tid >> 5;
    const int lane16 = lane & 15;
    const int hi     = lane >> 4;

    const int mBlock = blockIdx.y * BLK_M;
    const int nBlock = blockIdx.x * BLK_N;
    const int wm = (waveId >> 1) * WAVE_M;   // wave row offset inside block tile
    const int wn = (waveId & 1) * WAVE_N;    // wave col offset inside block tile

    // Tile origins in global memory (K advances inside the loop).
    const __bf16* gA = xb + (size_t)mBlock * KDIM;
    const __bf16* gB = wb + (size_t)nBlock * KDIM;

    f32x8 acc[4][4];
#pragma unroll
    for (int mi = 0; mi < 4; ++mi)
#pragma unroll
        for (int ni = 0; ni < 4; ++ni)
            acc[mi][ni] = (f32x8){};

    const int nKB = KDIM / BLK_K;   // 64 stages

#if HAS_TDM
    // Runtime scalar: low 32 bits of the generic address = LDS byte offset.
    const unsigned ldsBase = (unsigned)(unsigned long long)(void*)&smem[0];
    if (waveId == 0) {
        tdm_load_tile(gA + 0,     ldsBase + 0 * LDS_TILE_BYTES, MDIM);  // A stage 0
        tdm_load_tile(gB + 0,     ldsBase + 2 * LDS_TILE_BYTES, NDIM);  // B stage 0
        tdm_load_tile(gA + BLK_K, ldsBase + 1 * LDS_TILE_BYTES, MDIM);  // A stage 1
        tdm_load_tile(gB + BLK_K, ldsBase + 3 * LDS_TILE_BYTES, NDIM);  // B stage 1
    }
#endif

    for (int kb = 0; kb < nKB; ++kb) {
        const int st = kb & 1;
        const unsigned offA = (unsigned)st * LDS_TILE_BYTES;
        const unsigned offB = 2u * LDS_TILE_BYTES + (unsigned)st * LDS_TILE_BYTES;

#if HAS_TDM
        if (waveId == 0) {
            // Stage st's 2 TDM ops complete when <= (in-flight next stage) remain.
            if (kb == nKB - 1) __builtin_amdgcn_s_wait_tensorcnt(0);
            else               __builtin_amdgcn_s_wait_tensorcnt(2);
        }
        __syncthreads();
#else
        copy_tile_fallback(gA + (size_t)kb * BLK_K, (__bf16*)(smem + offA), tid);
        copy_tile_fallback(gB + (size_t)kb * BLK_K, (__bf16*)(smem + offB), tid);
        __syncthreads();
#endif

        const __bf16* As = (const __bf16*)(smem + offA);
        const __bf16* Bs = (const __bf16*)(smem + offB);

#pragma unroll
        for (int kk = 0; kk < BLK_K; kk += 32) {
            Bf16Frag af[4];
#pragma unroll
            for (int mi = 0; mi < 4; ++mi) {
                const int row = wm + mi * 16 + lane16;
                const __bf16* ap = As + row * LDS_ROW_ELEMS + kk + hi * 16;
                af[mi].h[0] = *(const bf16x8*)ap;          // K = kk + hi*16 .. +7
                af[mi].h[1] = *(const bf16x8*)(ap + 8);    // K = kk+hi*16+8 .. +15
            }
            Bf16Frag bf[4];
#pragma unroll
            for (int ni = 0; ni < 4; ++ni) {
                const int row = wn + ni * 16 + lane16;
                bf[ni].v = *(const bf16x16*)(Bs + row * LDS_ROW_ELEMS + kk + hi * 16);
            }
#pragma unroll
            for (int mi = 0; mi < 4; ++mi)
#pragma unroll
                for (int ni = 0; ni < 4; ++ni)
                    acc[mi][ni] = __builtin_amdgcn_wmma_f32_16x16x32_bf16(
                        false, af[mi].v, false, bf[ni].v,
                        (short)0, acc[mi][ni], false, false);
        }

        __syncthreads();   // all waves done reading stage before overwrite

#if HAS_TDM
        const int knext = (kb + 2) * BLK_K;
        if (waveId == 0 && knext < KDIM) {
            tdm_load_tile(gA + knext, ldsBase + offA, MDIM);
            tdm_load_tile(gB + knext, ldsBase + offB, NDIM);
        }
#endif
    }

    // Epilogue: bias add + store.
#pragma unroll
    for (int ni = 0; ni < 4; ++ni) {
        const int n = nBlock + wn + ni * 16 + lane16;
        const float bv = bias[n];
#pragma unroll
        for (int mi = 0; mi < 4; ++mi) {
            const int mrow = mBlock + wm + mi * 16 + hi * 8;
#pragma unroll
            for (int r = 0; r < 8; ++r)
                out[(size_t)(mrow + r) * NDIM + n] = acc[mi][ni][r] + bv;
        }
    }
}

// ---------------------------------------------------------------------------
// Launcher. Inputs: x(f32), centroids(f32), bias(f32), indices(int64).
// Workspace: [0,32MB) bf16 weight; [32MB,96MB) bf16 x.
// ---------------------------------------------------------------------------
extern "C" void kernel_launch(void* const* d_in, const int* in_sizes, int n_in,
                              void* d_out, int out_size, void* d_ws, size_t ws_size,
                              hipStream_t stream) {
    const float*     x         = (const float*)d_in[0];
    const float*     centroids = (const float*)d_in[1];
    const float*     bias      = (const float*)d_in[2];
    const long long* indices   = (const long long*)d_in[3];
    float*           out       = (float*)d_out;

    __bf16* wb = (__bf16*)d_ws;
    __bf16* xb = (__bf16*)((char*)d_ws + (size_t)NDIM * KDIM * sizeof(__bf16));

    {   // 1) Dequantize weights: 16M codes, 8 per thread.
        const size_t total = (size_t)NDIM * KDIM;
        wsl_dequant_weight<<<dim3((unsigned)(total / (8 * 256))), dim3(256), 0, stream>>>(
            indices, centroids, wb);
    }
    {   // 2) Convert activations: 32M floats, 8 per thread.
        const size_t total = (size_t)MDIM * KDIM;
        wsl_convert_x<<<dim3((unsigned)(total / (8 * 256))), dim3(256), 0, stream>>>(x, xb);
    }
    {   // 3) TDM-staged WMMA GEMM + bias.
        dim3 block(128);                          // 4 waves
        dim3 grid(NDIM / BLK_N, MDIM / BLK_M);    // 32 x 64
        wsl_wmma_gemm<<<grid, block, 0, stream>>>(xb, wb, bias, out);
    }
}
